// SpectraLoRALayer_39831526703187
// MI455X (gfx1250) — matrix operand order repaired and checked by reading, not compiled, not run on hardware
//
#include <hip/hip_runtime.h>

// ---------------------------------------------------------------------------
// SpectraLoRA fused layer for gfx1250 (MI455X), wave32 + WMMA bf16 +
// async global->LDS copies (ASYNCcnt path).
//
// Strategy: fold LoRA into a per-batch effective weight
//   Weff[b] = W + SCALING * sum_e g[b,e] * (Bm[e] @ A[e])      (K = E*R = 128)
// then a single big GEMM per batch: out[b] = x[b] @ Weff[b]^T + bias.
// Both GEMMs run on v_wmma_f32_16x16x32_bf16; the main GEMM feeds LDS with
// global_load_async_to_lds_b128 (double-buffered, s_wait_asynccnt).
// ---------------------------------------------------------------------------

typedef __bf16 bf16;
typedef __attribute__((ext_vector_type(16))) __bf16 v16bf;
typedef __attribute__((ext_vector_type(8)))  __bf16 v8bf;
typedef __attribute__((ext_vector_type(4)))  __bf16 v4bf;
typedef __attribute__((ext_vector_type(8)))  float  v8f;
typedef __attribute__((ext_vector_type(4)))  float  v4f;

#define IN_F   768
#define OUT_F  2304
#define NE     8
#define NR     16
#define ER     128          // E*R, rank of fused update
#define SCALING 2.0f        // ALPHA/R = 32/16
#define GDIM   64
#define NB     8            // batch
#define NS     2048         // seq
#define NROWS  (NB*NS)      // 16384

static __device__ __forceinline__ v16bf cat16(v8bf lo, v8bf hi) {
    return __builtin_shufflevector(lo, hi, 0,1,2,3,4,5,6,7,8,9,10,11,12,13,14,15);
}

// CDNA5 async global->LDS copy (16B), tracked by ASYNCcnt.
// Operand order per VGLOBAL encoding: vdst = LDS byte address, vaddr = 64-bit
// global address, no saddr.  LDS byte address = low 32 bits of the generic
// pointer (LDS aperture: addr[31:0] is the DS offset).
static __device__ __forceinline__ void async_ld16(unsigned lds, unsigned long long gaddr) {
    asm volatile("global_load_async_to_lds_b128 %0, %1, off"
                 :: "v"(lds), "v"(gaddr) : "memory");
}
static __device__ __forceinline__ void wait_async0() {
    asm volatile("s_wait_asynccnt 0x0" ::: "memory");
}

// ---------------------------------------------------------------------------
// K1: gate = softmax(z @ Wg^T + bg) over experts.  [8 batches x 8 experts]
// ---------------------------------------------------------------------------
__global__ __launch_bounds__(64)
void gate_kernel(const float* __restrict__ z, const float* __restrict__ Wg,
                 const float* __restrict__ bg, float* __restrict__ gate) {
    __shared__ float lg[NB][NE];
    int t = threadIdx.x;           // 64 threads: b = t/8, e = t%8
    int b = t >> 3, e = t & 7;
    float s = bg[e];
    #pragma unroll 8
    for (int g = 0; g < GDIM; ++g) s += z[b * GDIM + g] * Wg[e * GDIM + g];
    lg[b][e] = s;
    __syncthreads();
    float m = lg[b][0];
    #pragma unroll
    for (int j = 1; j < NE; ++j) m = fmaxf(m, lg[b][j]);
    float sum = 0.f;
    #pragma unroll
    for (int j = 0; j < NE; ++j) sum += __expf(lg[b][j] - m);
    gate[b * NE + e] = __expf(lg[b][e] - m) / sum;
}

// ---------------------------------------------------------------------------
// K2: pack operands for the rank-128 update GEMM (bf16):
//   bcat [OUT_F][ER]       : bcat[o][e*16+r]   = Bm[e][o][r]
//   ascT [NB][IN_F][ER]    : ascT[b][i][e*16+r] = SCALING*g[b,e]*A[e][r][i]
// ---------------------------------------------------------------------------
__global__ __launch_bounds__(256)
void pack_kernel(const float* __restrict__ A, const float* __restrict__ Bm,
                 const float* __restrict__ gate,
                 bf16* __restrict__ bcat, bf16* __restrict__ ascT) {
    int id = blockIdx.x * 256 + threadIdx.x;
    const int NBC = OUT_F * ER;                 // 294912
    if (id < NBC) {
        int o = id >> 7, er = id & 127, e = er >> 4, r = er & 15;
        bcat[id] = (bf16)Bm[((size_t)e * OUT_F + o) * NR + r];
    } else {
        int id2 = id - NBC;
        if (id2 < NB * IN_F * ER) {             // 786432
            int b   = id2 / (IN_F * ER);
            int rem = id2 % (IN_F * ER);
            int i = rem >> 7, er = rem & 127, e = er >> 4, r = er & 15;
            float v = SCALING * gate[b * NE + e] *
                      A[((size_t)e * NR + r) * IN_F + i];
            ascT[id2] = (bf16)v;                // id2 == ((b*IN_F)+i)*ER+er
        }
    }
}

// ---------------------------------------------------------------------------
// K3: convert x fp32 -> bf16 (row-major [16384][768])
// ---------------------------------------------------------------------------
__global__ __launch_bounds__(256)
void cvt_kernel(const float* __restrict__ x, bf16* __restrict__ xb) {
    int id = blockIdx.x * 256 + threadIdx.x;    // 1 thread = 4 elements
    v4f v = *(const v4f*)(x + (size_t)id * 4);
    v4bf o;
    o[0] = (bf16)v[0]; o[1] = (bf16)v[1]; o[2] = (bf16)v[2]; o[3] = (bf16)v[3];
    *(v4bf*)(xb + (size_t)id * 4) = o;
}

// ---------------------------------------------------------------------------
// K4: Weff[b] = W + bcat @ ascT[b]^T   (M=OUT_F, N=IN_F, K=128), bf16 out.
// One 16x16 C tile per wave, K=128 -> 4 x v_wmma_f32_16x16x32_bf16.
// ---------------------------------------------------------------------------
__global__ __launch_bounds__(256)
void weff_kernel(const bf16* __restrict__ bcat, const bf16* __restrict__ ascT,
                 const float* __restrict__ W, bf16* __restrict__ weff) {
    const int wave = threadIdx.x >> 5, lane = threadIdx.x & 31;
    const int id  = blockIdx.x * 8 + wave;        // tile id, 8*144*48 total
    const int b   = id / (144 * 48);
    const int rem = id % (144 * 48);
    const int ot = rem / 48, it = rem % 48;
    const int l16 = lane & 15, hh = lane >> 4;

    const bf16* ap = bcat + (size_t)(ot * 16 + l16) * ER + hh * 8;
    const bf16* bp = ascT + ((size_t)b * IN_F + it * 16 + l16) * ER + hh * 8;

    v8f acc = {};
    #pragma unroll
    for (int k0 = 0; k0 < ER; k0 += 32) {
        v16bf af = cat16(*(const v8bf*)(ap + k0), *(const v8bf*)(ap + k0 + 16));
        v16bf bf_ = cat16(*(const v8bf*)(bp + k0), *(const v8bf*)(bp + k0 + 16));
        acc = __builtin_amdgcn_wmma_f32_16x16x32_bf16(
                  false, af, false, bf_, (short)0, acc, false, false);
    }
    // C layout: VGPR v, lanes 0-15 -> M=v, N=lane; lanes 16-31 -> M=v+8.
    const int o0 = ot * 16 + hh * 8;
    const int i  = it * 16 + l16;
    #pragma unroll
    for (int v = 0; v < 8; ++v) {
        const int o = o0 + v;
        float val = W[(size_t)o * IN_F + i] + acc[v];
        weff[((size_t)b * OUT_F + o) * IN_F + i] = (bf16)val;
    }
}

// ---------------------------------------------------------------------------
// K5: main GEMM  out[b] = x_bf[b] @ Weff[b]^T + bias  (fp32 out)
// WG tile 128x128, 8 waves (2 Mwaves x 4 Nwaves), wave tile 64x32 = 4x2 WMMA
// tiles, K-step 32, double-buffered LDS fed by global_load_async_to_lds_b128.
// LDS rows padded to 40 bf16 (80 B, 16B-aligned, 20-bank stride).
// ---------------------------------------------------------------------------
#define LDA 40
#define LDSBUF (128 * LDA)           // elements per buffer
__global__ __launch_bounds__(256)
void gemm_kernel(const bf16* __restrict__ xb,   // [16384][768]
                 const bf16* __restrict__ weff, // [8][2304][768]
                 const float* __restrict__ bias,
                 float* __restrict__ out) {     // [16384][2304]
    __shared__ bf16 As[2][LDSBUF];
    __shared__ bf16 Bs[2][LDSBUF];

    const int tid  = threadIdx.x;
    const int lane = tid & 31, wave = tid >> 5;
    const int waveM = wave >> 2, waveN = wave & 3;
    const int l16 = lane & 15, hh = lane >> 4;

    const int batch = blockIdx.z;
    const int mBase = blockIdx.y * 128;          // within batch
    const int nBase = blockIdx.x * 128;

    const bf16* xrow = xb   + ((size_t)batch * NS + mBase) * IN_F;
    const bf16* wrow = weff + ((size_t)batch * OUT_F + nBase) * IN_F;

    // per-thread staging: 2 chunks of 8 bf16 (16 B) for each of A and B tiles
    const int c0r = tid >> 2, c0k = tid & 3;           // rows 0..63
    const int c1r = c0r + 64, c1k = c0k;               // rows 64..127

    // LDS byte addresses (buffer 0); buffer 1 = +LDSBUF*2 bytes
    const unsigned lA0 = (unsigned)(size_t)&As[0][c0r * LDA + c0k * 8];
    const unsigned lA1 = (unsigned)(size_t)&As[0][c1r * LDA + c1k * 8];
    const unsigned lB0 = (unsigned)(size_t)&Bs[0][c0r * LDA + c0k * 8];
    const unsigned lB1 = (unsigned)(size_t)&Bs[0][c1r * LDA + c1k * 8];
    const unsigned bufBytes = LDSBUF * (unsigned)sizeof(bf16);

    // global byte addresses at k=0
    const unsigned long long gX0 = (unsigned long long)(xrow + (size_t)c0r * IN_F + c0k * 8);
    const unsigned long long gX1 = (unsigned long long)(xrow + (size_t)c1r * IN_F + c1k * 8);
    const unsigned long long gW0 = (unsigned long long)(wrow + (size_t)c0r * IN_F + c0k * 8);
    const unsigned long long gW1 = (unsigned long long)(wrow + (size_t)c1r * IN_F + c1k * 8);

    v8f acc[4][2];
    #pragma unroll
    for (int mt = 0; mt < 4; ++mt)
        #pragma unroll
        for (int nt = 0; nt < 2; ++nt) acc[mt][nt] = (v8f){};

    // prologue: async-copy k-step 0 into buffer 0
    async_ld16(lA0, gX0);
    async_ld16(lA1, gX1);
    async_ld16(lB0, gW0);
    async_ld16(lB1, gW1);
    wait_async0();
    __syncthreads();

    int buf = 0;
    const int KSTEPS = IN_F / 32;               // 24
    for (int ks = 0; ks < KSTEPS; ++ks) {
        const bool more = (ks + 1) < KSTEPS;
        if (more) {
            // async-copy next k-step into the other buffer (overlaps compute)
            const unsigned long long koff = 64ull * (unsigned)(ks + 1); // 32 bf16
            const unsigned nbB = (buf ^ 1) ? bufBytes : 0u;
            async_ld16(lA0 + nbB, gX0 + koff);
            async_ld16(lA1 + nbB, gX1 + koff);
            async_ld16(lB0 + nbB, gW0 + koff);
            async_ld16(lB1 + nbB, gW1 + koff);
        }

        // gather fragments from LDS (documented 16-bit A/B layouts)
        v16bf afrag[4], bfrag[2];
        #pragma unroll
        for (int mt = 0; mt < 4; ++mt) {
            const bf16* p = &As[buf][(waveM * 64 + mt * 16 + l16) * LDA + hh * 8];
            afrag[mt] = cat16(*(const v8bf*)p, *(const v8bf*)(p + 16));
        }
        #pragma unroll
        for (int nt = 0; nt < 2; ++nt) {
            const bf16* p = &Bs[buf][(waveN * 32 + nt * 16 + l16) * LDA + hh * 8];
            bfrag[nt] = cat16(*(const v8bf*)p, *(const v8bf*)(p + 16));
        }
        #pragma unroll
        for (int mt = 0; mt < 4; ++mt)
            #pragma unroll
            for (int nt = 0; nt < 2; ++nt)
                acc[mt][nt] = __builtin_amdgcn_wmma_f32_16x16x32_bf16(
                    false, afrag[mt], false, bfrag[nt], (short)0, acc[mt][nt],
                    false, false);

        if (more) wait_async0();   // own-wave async copies into next buffer done
        __syncthreads();           // all waves' copies visible
        buf ^= 1;
    }

    // epilogue: C tile layout -> fp32 global stores, + bias
    #pragma unroll
    for (int nt = 0; nt < 2; ++nt) {
        const int col = nBase + waveN * 32 + nt * 16 + l16;
        const float bv = bias[col];
        #pragma unroll
        for (int mt = 0; mt < 4; ++mt) {
            const int row = mBase + waveM * 64 + mt * 16 + hh * 8;
            float* op = out + ((size_t)batch * NS + row) * OUT_F + col;
            v8f a = acc[mt][nt];
            #pragma unroll
            for (int v = 0; v < 8; ++v)
                op[(size_t)v * OUT_F] = a[v] + bv;
        }
    }
}

// ---------------------------------------------------------------------------
extern "C" void kernel_launch(void* const* d_in, const int* in_sizes, int n_in,
                              void* d_out, int out_size, void* d_ws, size_t ws_size,
                              hipStream_t stream) {
    const float* x    = (const float*)d_in[0];
    const float* z    = (const float*)d_in[1];
    const float* W    = (const float*)d_in[2];
    const float* bias = (const float*)d_in[3];
    const float* A    = (const float*)d_in[4];
    const float* Bm   = (const float*)d_in[5];
    const float* Wg   = (const float*)d_in[6];
    const float* bg   = (const float*)d_in[7];
    float* out = (float*)d_out;

    // workspace layout (bytes), all 16B-aligned
    char* ws = (char*)d_ws;
    float* gate = (float*)(ws);                               // 256 B
    bf16*  bcat = (bf16*)(ws + 1024);                         // 589,824 B
    bf16*  ascT = (bf16*)(ws + 1024 + 589824);                // 1,572,864 B
    bf16*  xbf  = (bf16*)(ws + 1024 + 589824 + 1572864);      // 25,165,824 B
    bf16*  weff = (bf16*)(ws + 1024 + 589824 + 1572864 + 25165824); // 28,311,552 B

    gate_kernel<<<1, 64, 0, stream>>>(z, Wg, bg, gate);

    {   // pack bcat + ascT: 294912 + 786432 = 1,081,344 threads
        const int total = OUT_F * ER + NB * IN_F * ER;
        pack_kernel<<<(total + 255) / 256, 256, 0, stream>>>(A, Bm, gate, bcat, ascT);
    }
    {   // x fp32 -> bf16, 4 elems/thread
        const int n4 = NROWS * IN_F / 4;                      // 3,145,728
        cvt_kernel<<<n4 / 256, 256, 0, stream>>>(x, xbf);
    }
    {   // Weff build: 8*144*48 = 55,296 tiles, 8 waves per block
        weff_kernel<<<(NB * 144 * 48) / 8, 256, 0, stream>>>(bcat, ascT, W, weff);
    }
    {   // main GEMM: grid (N tiles, M tiles per batch, batch)
        dim3 grid(OUT_F / 128, NS / 128, NB);
        gemm_kernel<<<grid, 256, 0, stream>>>(xbf, weff, bias, out);
    }
}
